// MixtralDecoderLayer_6073083756873
// MI455X (gfx1250) — matrix operand order repaired and checked by reading, not compiled, not run on hardware
//
#include <hip/hip_runtime.h>

// ---------------- problem constants (from reference) ----------------
constexpr int S   = 2048;
constexpr int H   = 1024;
constexpr int NH  = 16;
constexpr int NKV = 4;
constexpr int HD  = 64;
constexpr int ISZ = 3584;
constexpr int E   = 8;
constexpr int GR  = 2 * S;   // total gathered rows (TOPK * tokens)

#define DEV_INLINE __device__ __forceinline__

typedef __bf16 bf16;
typedef __attribute__((ext_vector_type(16))) __bf16 v16bf;
typedef __attribute__((ext_vector_type(8)))  __bf16 v8bf;
typedef __attribute__((ext_vector_type(8)))  float  v8f;

DEV_INLINE bf16 f2bf(float f) { return (bf16)f; }

DEV_INLINE v8f zero8() {
  v8f z;
#pragma unroll
  for (int j = 0; j < 8; ++j) z[j] = 0.f;
  return z;
}

DEV_INLINE v8f wmma_bf16(v16bf a, v16bf b, v8f c) {
  // D = A(16x32 bf16) x B(32x16 bf16) + C(16x16 f32)
  return __builtin_amdgcn_wmma_f32_16x16x32_bf16(false, a, false, b, (short)0, c,
                                                 false, false);
}

// A-matrix fragment (16x32 bf16, row-major source). lane owns row; k-map:
// k = k0 + 16*hi + j. Two contiguous 16B loads.
DEV_INLINE v16bf load_a_frag(const bf16* base, int ld, int row, int k0, int lane) {
  int hi = (lane >> 4) & 1;
  const bf16* p = base + (size_t)row * ld + k0 + 16 * hi;
  v8bf lo = *(const v8bf*)p;
  v8bf hv = *(const v8bf*)(p + 8);
  v16bf r;
#pragma unroll
  for (int j = 0; j < 8; ++j) { r[j] = lo[j]; r[j + 8] = hv[j]; }
  return r;
}

// B-matrix fragment (32x16 bf16). lane owns column n; k-map: k = k0 + 16*hi + j.
// Source laid out [n][k] (k contiguous), so one 32B contiguous run per lane.
DEV_INLINE v16bf load_b_frag(const bf16* base, int ld, int ncol, int k0, int lane) {
  int hi = (lane >> 4) & 1;
  const bf16* p = base + (size_t)ncol * ld + k0 + 16 * hi;
  v8bf lo = *(const v8bf*)p;
  v8bf hv = *(const v8bf*)(p + 8);
  v16bf r;
#pragma unroll
  for (int j = 0; j < 8; ++j) { r[j] = lo[j]; r[j + 8] = hv[j]; }
  return r;
}

DEV_INLINE void lds_fence() {
  // CDNA5 split counter: wait for outstanding DS ops, block reordering.
  __asm__ volatile("s_wait_dscnt 0" ::: "memory");
}

// ---------------- RMSNorm (fp32 in -> bf16 out) ----------------
__global__ __launch_bounds__(256) void rmsnorm_kernel(const float* __restrict__ x,
                                                      const float* __restrict__ w,
                                                      bf16* __restrict__ out) {
  int row = blockIdx.x;
  int tid = threadIdx.x, lane = tid & 31, wave = tid >> 5;
  __shared__ float red[8];
  const float* xr = x + (size_t)row * H;
  float ss = 0.f;
  for (int j = tid; j < H; j += 256) { float v = xr[j]; ss += v * v; }
#pragma unroll
  for (int off = 16; off > 0; off >>= 1) ss += __shfl_xor(ss, off, 32);
  if (lane == 0) red[wave] = ss;
  __syncthreads();
  float tot = 0.f;
#pragma unroll
  for (int i = 0; i < 8; ++i) tot += red[i];
  float inv = rsqrtf(tot / (float)H + 1e-5f);
  bf16* orow = out + (size_t)row * H;
  for (int j = tid; j < H; j += 256) orow[j] = f2bf(xr[j] * inv * w[j]);
}

// ---------------- generic WMMA GEMM: C[M,N] = A_bf16[M,K] @ B_f32->bf16[K,N] ----
// block = 256 threads = 8 waves (4 M x 2 N), block tile 64x64, K step 32.
template <bool OUT_BF16, bool RES>
__global__ __launch_bounds__(256) void gemm_kernel(
    const bf16* __restrict__ A, const float* __restrict__ B,
    float* __restrict__ Cf, bf16* __restrict__ Cb,
    const float* __restrict__ resid, int M, int N, int K) {
  __shared__ __align__(32) bf16 ldsB[64 * 32];  // [n][k] transposed tile
  int tid = threadIdx.x, lane = tid & 31, wave = tid >> 5;
  int wm = wave & 3, wn = wave >> 2;
  int bm = blockIdx.x * 64, bn = blockIdx.y * 64;
  int mrow = bm + wm * 16 + (lane & 15);
  if (mrow >= M) mrow = M - 1;
  v8f acc0 = zero8(), acc1 = zero8();
  for (int k0 = 0; k0 < K; k0 += 32) {
    __syncthreads();
#pragma unroll
    for (int i = 0; i < 8; ++i) {
      int idx = tid + i * 256;          // 2048 elements
      int n = idx & 63, kk = idx >> 6;  // coalesced over n
      ldsB[n * 32 + kk] = f2bf(B[(size_t)(k0 + kk) * N + bn + n]);
    }
    __syncthreads();
    if (k0 + 32 < K) __builtin_prefetch(&B[(size_t)(k0 + 32) * N + bn + (tid & 63)], 0, 1);
    v16bf a = load_a_frag(A, K, mrow, k0, lane);
    int nb = wn * 32 + (lane & 15);
    v16bf b0 = load_b_frag(ldsB, 32, nb, 0, lane);
    acc0 = wmma_bf16(a, b0, acc0);
    v16bf b1 = load_b_frag(ldsB, 32, nb + 16, 0, lane);
    acc1 = wmma_bf16(a, b1, acc1);
  }
  int hi = (lane >> 4) & 1, col = lane & 15;
#pragma unroll
  for (int r = 0; r < 8; ++r) {
    int row = bm + wm * 16 + r + 8 * hi;
    if (row < M) {
      size_t o0 = (size_t)row * N + bn + wn * 32 + col;
      size_t o1 = o0 + 16;
      if (OUT_BF16) {
        Cb[o0] = f2bf(acc0[r]);
        Cb[o1] = f2bf(acc1[r]);
      } else {
        float r0 = acc0[r], r1 = acc1[r];
        if (RES) { r0 += resid[o0]; r1 += resid[o1]; }
        Cf[o0] = r0;
        Cf[o1] = r1;
      }
    }
  }
}

// ---------------- RoPE (fp32 in -> bf16 out), scale folded for Q ----------------
__global__ __launch_bounds__(256) void rope_kernel(
    const float* __restrict__ in, const int* __restrict__ pos_ids,
    bf16* __restrict__ out, int nh, float scale) {
  int idx = blockIdx.x * 256 + threadIdx.x;
  if (idx >= S * nh * HD) return;
  int d = idx & 63;
  int s = idx / (nh * HD);
  float pos = (float)pos_ids[s];
  int i = d & 31;
  float invf = __expf(-(2.0f * (float)i / 64.0f) * __logf(1.0e6f));
  float f = pos * invf;
  float c = __cosf(f), sn = __sinf(f);
  float x = in[idx];
  float xp = (d < 32) ? -in[idx + 32] : in[idx - 32];
  out[idx] = f2bf((x * c + xp * sn) * scale);
}

// ---------------- flash attention (causal, GQA) ----------------
// grid (NH, S/64), block 128 (4 waves); wave owns 16 query rows.
// Key-block loop is block-uniform -> K/V tiles staged cooperatively in LDS.
__global__ __launch_bounds__(128, 1) void attn_kernel(
    const bf16* __restrict__ Q, const bf16* __restrict__ Kmat,
    const bf16* __restrict__ V, bf16* __restrict__ O) {
  __shared__ __align__(32) bf16 ldsP[4][16 * 32];  // per-wave P tile (C->A relayout)
  __shared__ __align__(32) bf16 ldsK[32 * 64];     // [key][hd]
  __shared__ __align__(32) bf16 ldsV[64 * 32];     // [hd][key] (transposed)
  int h = blockIdx.x, qb = blockIdx.y;
  int lane = threadIdx.x & 31, wave = threadIdx.x >> 5;
  int kvh = h >> 2;  // NH/NKV = 4
  int qr0 = qb * 64 + wave * 16;
  int mrow = qr0 + (lane & 15);
  const bf16* Qh = Q + h * HD;
  const bf16* Kh = Kmat + kvh * HD;
  const bf16* Vh = V + kvh * HD;
  v16bf qa0 = load_a_frag(Qh, NH * HD, mrow, 0, lane);
  v16bf qa1 = load_a_frag(Qh, NH * HD, mrow, 32, lane);
  float mst[8], lst[8];
#pragma unroll
  for (int r = 0; r < 8; ++r) { mst[r] = -1e30f; lst[r] = 0.f; }
  v8f o0 = zero8(), o1 = zero8(), o2 = zero8(), o3 = zero8();
  int hi = (lane >> 4) & 1, col = lane & 15;
  int kb_end = 2 * qb + 1;  // block-uniform causal extent
  for (int kb = 0; kb <= kb_end; ++kb) {
    int kbase = kb * 32;
    // ---- cooperative K/V tile staging (32 keys x 64 hd, bf16) ----
    __syncthreads();  // previous iteration's LDS reads complete
#pragma unroll
    for (int i = 0; i < 2; ++i) {
      int c = threadIdx.x + i * 128;        // 256 chunks of 8 halfs
      int key = c >> 3, hd0 = (c & 7) * 8;
      const bf16* kp = Kh + (size_t)(kbase + key) * (NKV * HD) + hd0;
      const bf16* vp = Vh + (size_t)(kbase + key) * (NKV * HD) + hd0;
      v8bf kv = *(const v8bf*)kp;
      v8bf vv = *(const v8bf*)vp;
      *(v8bf*)(ldsK + key * 64 + hd0) = kv;  // row-major: B-frag for QK^T
#pragma unroll
      for (int j = 0; j < 8; ++j) ldsV[(hd0 + j) * 32 + key] = vv[j];  // transposed
    }
    __syncthreads();
    // ---- scores: S = Q(16x64) @ K^T(64x32) via 4 WMMAs ----
    v8f s0 = zero8(), s1 = zero8();
    {
      v16bf kf = load_b_frag(ldsK, 64, col, 0, lane);
      s0 = wmma_bf16(qa0, kf, s0);
      kf = load_b_frag(ldsK, 64, col, 32, lane);
      s0 = wmma_bf16(qa1, kf, s0);
      kf = load_b_frag(ldsK, 64, col + 16, 0, lane);
      s1 = wmma_bf16(qa0, kf, s1);
      kf = load_b_frag(ldsK, 64, col + 16, 32, lane);
      s1 = wmma_bf16(qa1, kf, s1);
    }
    // ---- causal mask + online softmax ----
    int c0 = kbase + col, c1 = c0 + 16;
#pragma unroll
    for (int r = 0; r < 8; ++r) {
      int qr = qr0 + r + 8 * hi;
      float v0 = (c0 <= qr) ? s0[r] : -1e30f;
      float v1 = (c1 <= qr) ? s1[r] : -1e30f;
      float mx = fmaxf(v0, v1);
#pragma unroll
      for (int off = 1; off < 16; off <<= 1) mx = fmaxf(mx, __shfl_xor(mx, off, 32));
      float mnew = fmaxf(mst[r], mx);
      float p0 = __expf(v0 - mnew);
      float p1 = __expf(v1 - mnew);
      float rs = p0 + p1;
#pragma unroll
      for (int off = 1; off < 16; off <<= 1) rs += __shfl_xor(rs, off, 32);
      float fs = __expf(mst[r] - mnew);
      lst[r] = lst[r] * fs + rs;
      mst[r] = mnew;
      o0[r] *= fs; o1[r] *= fs; o2[r] *= fs; o3[r] *= fs;
      int rr = r + 8 * hi;
      ldsP[wave][rr * 32 + col] = f2bf(p0);
      ldsP[wave][rr * 32 + 16 + col] = f2bf(p1);
    }
    lds_fence();  // C-layout stores -> A-layout reads (same wave, cross-lane)
    v16bf pa = load_a_frag(&ldsP[wave][0], 32, col, 0, lane);
    // ---- O += P(16x32) @ V(32x64) via 4 WMMAs, V frags from transposed LDS ----
    {
      v16bf vb = load_b_frag(ldsV, 32, col, 0, lane);
      o0 = wmma_bf16(pa, vb, o0);
      vb = load_b_frag(ldsV, 32, col + 16, 0, lane);
      o1 = wmma_bf16(pa, vb, o1);
      vb = load_b_frag(ldsV, 32, col + 32, 0, lane);
      o2 = wmma_bf16(pa, vb, o2);
      vb = load_b_frag(ldsV, 32, col + 48, 0, lane);
      o3 = wmma_bf16(pa, vb, o3);
    }
  }
#pragma unroll
  for (int r = 0; r < 8; ++r) {
    int qr = qr0 + r + 8 * hi;
    float inv = 1.0f / lst[r];
    size_t base = (size_t)qr * (NH * HD) + h * HD;
    O[base + col]      = f2bf(o0[r] * inv);
    O[base + 16 + col] = f2bf(o1[r] * inv);
    O[base + 32 + col] = f2bf(o2[r] * inv);
    O[base + 48 + col] = f2bf(o3[r] * inv);
  }
}

// ---------------- MoE gating: softmax + top-2 + counts ----------------
__global__ __launch_bounds__(256) void gate_kernel(
    const bf16* __restrict__ xn, const float* __restrict__ gw,
    int* __restrict__ top_i, float* __restrict__ top_w, int* __restrict__ counts) {
  int lane = threadIdx.x & 31, wave = threadIdx.x >> 5;
  int t = blockIdx.x * 8 + wave;
  float acc[E];
#pragma unroll
  for (int e = 0; e < E; ++e) acc[e] = 0.f;
  const bf16* xr = xn + (size_t)t * H;
  for (int j = lane; j < H; j += 32) {
    float xv = (float)xr[j];
#pragma unroll
    for (int e = 0; e < E; ++e) acc[e] += xv * gw[j * E + e];
  }
#pragma unroll
  for (int e = 0; e < E; ++e) {
#pragma unroll
    for (int off = 16; off > 0; off >>= 1) acc[e] += __shfl_xor(acc[e], off, 32);
  }
  if (lane == 0) {
    float mx = acc[0];
    for (int e = 1; e < E; ++e) mx = fmaxf(mx, acc[e]);
    float p[E];
    float sum = 0.f;
    for (int e = 0; e < E; ++e) { p[e] = __expf(acc[e] - mx); sum += p[e]; }
    for (int e = 0; e < E; ++e) p[e] /= sum;
    int a1 = 0;
    for (int e = 1; e < E; ++e) if (p[e] > p[a1]) a1 = e;
    int a2 = (a1 == 0) ? 1 : 0;
    for (int e = 0; e < E; ++e) if (e != a1 && p[e] > p[a2]) a2 = e;
    float norm = p[a1] + p[a2];
    top_i[t * 2] = a1;  top_i[t * 2 + 1] = a2;
    top_w[t * 2] = p[a1] / norm;  top_w[t * 2 + 1] = p[a2] / norm;
    atomicAdd(&counts[a1], 1);
    atomicAdd(&counts[a2], 1);
  }
}

__global__ void zero_counts_kernel(int* counts) {
  if (threadIdx.x < E) counts[threadIdx.x] = 0;
}

__global__ void scan_kernel(const int* __restrict__ counts, int* __restrict__ offsets,
                            int* __restrict__ cursor) {
  if (threadIdx.x == 0) {
    int o = 0;
    for (int e = 0; e < E; ++e) { offsets[e] = o; cursor[e] = o; o += counts[e]; }
  }
}

__global__ void fill_kernel(const int* __restrict__ top_i, int* __restrict__ cursor,
                            int* __restrict__ rowidx, int* __restrict__ slot) {
  int t = blockIdx.x * 256 + threadIdx.x;
  if (t >= S) return;
#pragma unroll
  for (int kk = 0; kk < 2; ++kk) {
    int e = top_i[t * 2 + kk];
    int pos = atomicAdd(&cursor[e], 1);
    rowidx[pos] = t;
    slot[t * 2 + kk] = pos;
  }
}

// ---------------- grouped MoE GEMM1: Hb = silu(X@w1[e]) * (X@w3[e]) ----------------
// grid (E, S/64, ISZ/64); early-exit beyond per-expert count.
__global__ __launch_bounds__(256) void moe_gemm1_kernel(
    const bf16* __restrict__ X, const float* __restrict__ W1,
    const float* __restrict__ W3, bf16* __restrict__ Hb,
    const int* __restrict__ rowidx, const int* __restrict__ counts,
    const int* __restrict__ offsets) {
  __shared__ __align__(32) bf16 lds1[64 * 32];
  __shared__ __align__(32) bf16 lds3[64 * 32];
  int e = blockIdx.x;
  int cnt = counts[e], off = offsets[e];
  int bm = blockIdx.y * 64;
  if (bm >= cnt) return;
  int bn = blockIdx.z * 64;
  int tid = threadIdx.x, lane = tid & 31, wave = tid >> 5;
  int wm = wave & 3, wn = wave >> 2;
  int mloc = bm + wm * 16 + (lane & 15);
  if (mloc >= cnt) mloc = cnt - 1;
  int tok = rowidx[off + mloc];  // gathered token row
  const float* B1 = W1 + (size_t)e * H * ISZ;
  const float* B3 = W3 + (size_t)e * H * ISZ;
  v8f a10 = zero8(), a11 = zero8(), a30 = zero8(), a31 = zero8();
  for (int k0 = 0; k0 < H; k0 += 32) {
    __syncthreads();
#pragma unroll
    for (int i = 0; i < 8; ++i) {
      int idx = tid + i * 256;
      int n = idx & 63, kk = idx >> 6;
      size_t g = (size_t)(k0 + kk) * ISZ + bn + n;
      lds1[n * 32 + kk] = f2bf(B1[g]);
      lds3[n * 32 + kk] = f2bf(B3[g]);
    }
    __syncthreads();
    if (k0 + 32 < H) __builtin_prefetch(&B1[(size_t)(k0 + 32) * ISZ + bn + (tid & 63)], 0, 1);
    v16bf a = load_a_frag(X, H, tok, k0, lane);
    int nb = wn * 32 + (lane & 15);
    v16bf b10 = load_b_frag(lds1, 32, nb, 0, lane);
    a10 = wmma_bf16(a, b10, a10);
    v16bf b11 = load_b_frag(lds1, 32, nb + 16, 0, lane);
    a11 = wmma_bf16(a, b11, a11);
    v16bf b30 = load_b_frag(lds3, 32, nb, 0, lane);
    a30 = wmma_bf16(a, b30, a30);
    v16bf b31 = load_b_frag(lds3, 32, nb + 16, 0, lane);
    a31 = wmma_bf16(a, b31, a31);
  }
  int hi = (lane >> 4) & 1, colc = lane & 15;
#pragma unroll
  for (int r = 0; r < 8; ++r) {
    int row = bm + wm * 16 + r + 8 * hi;
    if (row < cnt) {
      size_t o0 = (size_t)(off + row) * ISZ + bn + wn * 32 + colc;
      float g0 = a10[r], g1 = a11[r];
      Hb[o0]      = f2bf(g0 / (1.f + __expf(-g0)) * a30[r]);
      Hb[o0 + 16] = f2bf(g1 / (1.f + __expf(-g1)) * a31[r]);
    }
  }
}

// ---------------- grouped MoE GEMM2: Gout = Hb @ w2[e] ----------------
__global__ __launch_bounds__(256) void moe_gemm2_kernel(
    const bf16* __restrict__ Hb, const float* __restrict__ W2,
    float* __restrict__ Gout, const int* __restrict__ counts,
    const int* __restrict__ offsets) {
  __shared__ __align__(32) bf16 ldsB[64 * 32];
  int e = blockIdx.x;
  int cnt = counts[e], off = offsets[e];
  int bm = blockIdx.y * 64;
  if (bm >= cnt) return;
  int bn = blockIdx.z * 64;
  int tid = threadIdx.x, lane = tid & 31, wave = tid >> 5;
  int wm = wave & 3, wn = wave >> 2;
  int mloc = bm + wm * 16 + (lane & 15);
  if (mloc >= cnt) mloc = cnt - 1;
  int arow = off + mloc;
  const float* B = W2 + (size_t)e * ISZ * H;
  v8f acc0 = zero8(), acc1 = zero8();
  for (int k0 = 0; k0 < ISZ; k0 += 32) {
    __syncthreads();
#pragma unroll
    for (int i = 0; i < 8; ++i) {
      int idx = tid + i * 256;
      int n = idx & 63, kk = idx >> 6;
      ldsB[n * 32 + kk] = f2bf(B[(size_t)(k0 + kk) * H + bn + n]);
    }
    __syncthreads();
    if (k0 + 32 < ISZ) __builtin_prefetch(&B[(size_t)(k0 + 32) * H + bn + (tid & 63)], 0, 1);
    v16bf a = load_a_frag(Hb, ISZ, arow, k0, lane);
    int nb = wn * 32 + (lane & 15);
    v16bf b0 = load_b_frag(ldsB, 32, nb, 0, lane);
    acc0 = wmma_bf16(a, b0, acc0);
    v16bf b1 = load_b_frag(ldsB, 32, nb + 16, 0, lane);
    acc1 = wmma_bf16(a, b1, acc1);
  }
  int hi = (lane >> 4) & 1, colc = lane & 15;
#pragma unroll
  for (int r = 0; r < 8; ++r) {
    int row = bm + wm * 16 + r + 8 * hi;
    if (row < cnt) {
      size_t o0 = (size_t)(off + row) * H + bn + wn * 32 + colc;
      Gout[o0] = acc0[r];
      Gout[o0 + 16] = acc1[r];
    }
  }
}

// ---------------- finalize: out = h + w0*gout[slot0] + w1*gout[slot1] ----------------
__global__ __launch_bounds__(256) void finalize_kernel(
    const float* __restrict__ h, const float* __restrict__ gout,
    const int* __restrict__ slot, const float* __restrict__ top_w,
    float* __restrict__ out) {
  int idx = blockIdx.x * 256 + threadIdx.x;
  if (idx >= S * H) return;
  int t = idx >> 10;        // H == 1024
  int j = idx & (H - 1);
  int s0 = slot[t * 2], s1 = slot[t * 2 + 1];
  out[idx] = h[idx] + top_w[t * 2] * gout[(size_t)s0 * H + j] +
             top_w[t * 2 + 1] * gout[(size_t)s1 * H + j];
}

// ---------------- host launcher ----------------
extern "C" void kernel_launch(void* const* d_in, const int* in_sizes, int n_in,
                              void* d_out, int out_size, void* d_ws, size_t ws_size,
                              hipStream_t stream) {
  (void)in_sizes; (void)n_in; (void)out_size; (void)ws_size;
  const float* hidden  = (const float*)d_in[0];
  const int*   pos     = (const int*)d_in[1];
  const float* wq      = (const float*)d_in[2];
  const float* wk      = (const float*)d_in[3];
  const float* wv      = (const float*)d_in[4];
  const float* wo      = (const float*)d_in[5];
  const float* in_nw   = (const float*)d_in[6];
  const float* post_nw = (const float*)d_in[7];
  const float* gw      = (const float*)d_in[8];
  const float* w1      = (const float*)d_in[9];
  const float* w3      = (const float*)d_in[10];
  const float* w2      = (const float*)d_in[11];
  float* out = (float*)d_out;

  // bump-allocate workspace (~80 MB total)
  char* ws = (char*)d_ws;
  size_t cur = 0;
  auto alloc = [&](size_t bytes) -> char* {
    char* p = ws + cur;
    cur = (cur + bytes + 255) & ~(size_t)255;
    return p;
  };
  bf16*  xn1    = (bf16*)alloc((size_t)S * H * 2);
  float* qf     = (float*)alloc((size_t)S * NH * HD * 4);
  float* kf     = (float*)alloc((size_t)S * NKV * HD * 4);
  bf16*  qbf    = (bf16*)alloc((size_t)S * NH * HD * 2);
  bf16*  kbf    = (bf16*)alloc((size_t)S * NKV * HD * 2);
  bf16*  vbf    = (bf16*)alloc((size_t)S * NKV * HD * 2);
  bf16*  att    = (bf16*)alloc((size_t)S * NH * HD * 2);
  float* hres   = (float*)alloc((size_t)S * H * 4);
  bf16*  xn2    = (bf16*)alloc((size_t)S * H * 2);
  int*   top_i  = (int*)alloc((size_t)S * 2 * 4);
  float* topw   = (float*)alloc((size_t)S * 2 * 4);
  int*   counts = (int*)alloc(E * 4);
  int*   offs   = (int*)alloc(E * 4);
  int*   cursor = (int*)alloc(E * 4);
  int*   rowidx = (int*)alloc((size_t)GR * 4);
  int*   slot   = (int*)alloc((size_t)S * 2 * 4);
  bf16*  hbuf   = (bf16*)alloc((size_t)GR * ISZ * 2);
  float* gout   = (float*)alloc((size_t)GR * H * 4);

  // 1) input norm
  rmsnorm_kernel<<<S, 256, 0, stream>>>(hidden, in_nw, xn1);
  // 2) Q/K/V projections (WMMA)
  gemm_kernel<false, false><<<dim3(S / 64, (NH * HD) / 64), 256, 0, stream>>>(
      xn1, wq, qf, nullptr, nullptr, S, NH * HD, H);
  gemm_kernel<false, false><<<dim3(S / 64, (NKV * HD) / 64), 256, 0, stream>>>(
      xn1, wk, kf, nullptr, nullptr, S, NKV * HD, H);
  gemm_kernel<true, false><<<dim3(S / 64, (NKV * HD) / 64), 256, 0, stream>>>(
      xn1, wv, nullptr, vbf, nullptr, S, NKV * HD, H);
  // 3) RoPE (scale 1/sqrt(HD) folded into Q)
  rope_kernel<<<(S * NH * HD) / 256, 256, 0, stream>>>(qf, pos, qbf, NH, 0.125f);
  rope_kernel<<<(S * NKV * HD) / 256, 256, 0, stream>>>(kf, pos, kbf, NKV, 1.0f);
  // 4) flash attention
  attn_kernel<<<dim3(NH, S / 64), 128, 0, stream>>>(qbf, kbf, vbf, att);
  // 5) output projection + residual -> hres
  gemm_kernel<false, true><<<dim3(S / 64, H / 64), 256, 0, stream>>>(
      att, wo, hres, nullptr, hidden, S, H, NH * HD);
  // 6) post norm
  rmsnorm_kernel<<<S, 256, 0, stream>>>(hres, post_nw, xn2);
  // 7) gating + token gather
  zero_counts_kernel<<<1, 32, 0, stream>>>(counts);
  gate_kernel<<<S / 8, 256, 0, stream>>>(xn2, gw, top_i, topw, counts);
  scan_kernel<<<1, 32, 0, stream>>>(counts, offs, cursor);
  fill_kernel<<<(S + 255) / 256, 256, 0, stream>>>(top_i, cursor, rowidx, slot);
  // 8) grouped expert GEMMs (WMMA)
  moe_gemm1_kernel<<<dim3(E, S / 64, ISZ / 64), 256, 0, stream>>>(
      xn2, w1, w3, hbuf, rowidx, counts, offs);
  moe_gemm2_kernel<<<dim3(E, S / 64, H / 64), 256, 0, stream>>>(
      hbuf, w2, gout, counts, offs);
  // 9) weighted combine + residual (deterministic, no float atomics)
  finalize_kernel<<<(S * H) / 256, 256, 0, stream>>>(hres, gout, slot, topw, out);
}
